// MGAT_12970801234014
// MI455X (gfx1250) — compile-verified, matrix operand order checked by reading, not run on hardware
//
#include <hip/hip_runtime.h>
#include <hip/hip_bf16.h>
#include <math.h>
#include <stdint.h>

typedef __attribute__((ext_vector_type(16))) _Float16 v16h;
typedef __attribute__((ext_vector_type(8)))  float    v8f;

#define NEG_SLOPE 0.2f
#define EPS_GN 1e-5f

// ---------- helpers ----------
__device__ __forceinline__ unsigned fenc(float f) {
    unsigned u = __float_as_uint(f);
    return (u >> 31) ? ~u : (u | 0x80000000u);  // monotone float->uint
}
__device__ __forceinline__ float fdec(unsigned e) {
    return __uint_as_float((e >> 31) ? (e & 0x7fffffffu) : ~e);
}
__device__ __forceinline__ float gelu_tanh(float x) {
    float x3 = x * x * x;
    return 0.5f * x * (1.f + tanhf(0.7978845608028654f * (x + 0.044715f * x3)));
}
__device__ __forceinline__ float wave_sum32(float p) {
    p += __shfl_xor(p, 1, 32); p += __shfl_xor(p, 2, 32);
    p += __shfl_xor(p, 4, 32); p += __shfl_xor(p, 8, 32);
    p += __shfl_xor(p, 16, 32);
    return p;
}

// ---- gfx1250 async copy global->LDS (ASYNCcnt path), inline asm per CDNA5 ISA ----
__device__ __forceinline__ void async_b128(void* lds_dst, const void* gsrc) {
    unsigned loff = (unsigned)(size_t)lds_dst;            // aperture low 32 bits == LDS byte offset
    asm volatile("global_load_async_to_lds_b128 %0, %1, off"
                 :: "v"(loff), "v"(gsrc) : "memory");
}
__device__ __forceinline__ void wait_async0() { asm volatile("s_wait_asynccnt 0" ::: "memory"); }
__device__ __forceinline__ void wait_async4() { asm volatile("s_wait_asynccnt 4" ::: "memory"); }

// build a 16-wide f16 A/B fragment (ISA 16x32 layout) from an f32 LDS row: K in
// {kb..kb+7, 16+kb..16+kb+7}; vectorized LDS reads + v_cvt_pk_f16_f32 pairs.
__device__ __forceinline__ void frag_from_lds_f32(v16h& f, const float* row, int kb) {
    const float4 p0 = *(const float4*)(row + kb);
    const float4 p1 = *(const float4*)(row + kb + 4);
    const float4 p2 = *(const float4*)(row + 16 + kb);
    const float4 p3 = *(const float4*)(row + 16 + kb + 4);
    f[0]  = (_Float16)p0.x; f[1]  = (_Float16)p0.y; f[2]  = (_Float16)p0.z; f[3]  = (_Float16)p0.w;
    f[4]  = (_Float16)p1.x; f[5]  = (_Float16)p1.y; f[6]  = (_Float16)p1.z; f[7]  = (_Float16)p1.w;
    f[8]  = (_Float16)p2.x; f[9]  = (_Float16)p2.y; f[10] = (_Float16)p2.z; f[11] = (_Float16)p2.w;
    f[12] = (_Float16)p3.x; f[13] = (_Float16)p3.y; f[14] = (_Float16)p3.z; f[15] = (_Float16)p3.w;
}

__global__ void zero_kernel(unsigned* __restrict__ p, long long n) {
    long long i = (long long)blockIdx.x * 256 + threadIdx.x;
    if (i < n) p[i] = 0u;
}

// ---------- generic tiled WMMA GEMM:  C = act(A[M,K] @ B[K,Ncols] + bias) ----------
// 256 thr = 8 waves; block tile 64x64; wave tile 16x32; K step 32.
// Double-buffered: async global->LDS copies for tile t+1 overlap WMMA on tile t.
#define LDA_F32 36   // f32 row stride for A tile (bank-conflict-free: gcd trick)
template <int ACT>
__global__ __launch_bounds__(256) void gemm_wmma(
    const float* __restrict__ A, const float* __restrict__ B,
    const float* __restrict__ bias, float* __restrict__ Cout,
    int M, int K, int Ncols)
{
    __shared__ float sA[2][64 * LDA_F32];
    __shared__ float sB[2][32 * 64];
    const int tid = threadIdx.x;
    const int lane = tid & 31, wv = tid >> 5;
    const int wm = wv & 3, wn = wv >> 2;           // 4 m-tiles x 2 n-groups
    const int mbase = blockIdx.x * 64, nbase = blockIdx.y * 64;
    const int lane15 = lane & 15, hi = lane >> 4, kb = hi * 8;

    // per-thread staging coordinates
    const int ar = tid >> 2, ac0 = (tid & 3) * 8;        // A: 64 rows x 32 cols
    const int br = tid >> 3, bc0 = (tid & 7) * 8;        // B: 32 rows x 64 cols
    const int agr = mbase + ar;
    const bool a_ok = (agr < M);

    auto stage = [&](int k0, int buf) {
        if (a_ok) {
            const float* ap = A + (size_t)agr * K + k0 + ac0;
            float* ad = &sA[buf][ar * LDA_F32 + ac0];
            async_b128(ad, ap);
            async_b128(ad + 4, ap + 4);
        }
        const float* bp = B + (size_t)(k0 + br) * Ncols + nbase + bc0;
        float* bd = &sB[buf][br * 64 + bc0];
        async_b128(bd, bp);
        async_b128(bd + 4, bp + 4);
    };

    v8f acc0 = {}; v8f acc1 = {};
    const int T = K >> 5;
    stage(0, 0);
    for (int t = 0; t < T; t++) {
        const int buf = t & 1;
        if (t + 1 < T) { stage((t + 1) << 5, buf ^ 1); wait_async4(); }
        else           { wait_async0(); }
        __syncthreads();

        v16h a;
        frag_from_lds_f32(a, &sA[buf][(wm * 16 + lane15) * LDA_F32], kb);
        v16h b0, b1;
        {
            const int nloc0 = wn * 32 + lane15;
            const float* sb = sB[buf];
#pragma unroll
            for (int j = 0; j < 8; j++) {
                b0[j]     = (_Float16)sb[(kb + j) * 64 + nloc0];
                b0[j + 8] = (_Float16)sb[(16 + kb + j) * 64 + nloc0];
                b1[j]     = (_Float16)sb[(kb + j) * 64 + nloc0 + 16];
                b1[j + 8] = (_Float16)sb[(16 + kb + j) * 64 + nloc0 + 16];
            }
        }
        acc0 = __builtin_amdgcn_wmma_f32_16x16x32_f16(false, a, false, b0, (short)0, acc0, false, false);
        acc1 = __builtin_amdgcn_wmma_f32_16x16x32_f16(false, a, false, b1, (short)0, acc1, false, false);
        __syncthreads();
    }

    // store: VGPR r -> M = r + hi*8 ; N = lane15 (+ tile base)
    const int mrow = mbase + wm * 16 + hi * 8;
    const int ng0 = nbase + wn * 32 + lane15;
    const int ng1 = ng0 + 16;
#pragma unroll
    for (int r = 0; r < 8; r++) {
        int mg = mrow + r;
        if (mg < M) {
            float v0 = acc0[r], v1 = acc1[r];
            if (bias) { v0 += bias[ng0]; v1 += bias[ng1]; }
            if (ACT == 1) { v0 = gelu_tanh(v0); v1 = gelu_tanh(v1); }
            Cout[(size_t)mg * Ncols + ng0] = v0;
            Cout[(size_t)mg * Ncols + ng1] = v1;
        }
    }
}

// ---------- mask = sigmoid(<h@Wm, instr[batch]>/sqrt(C));  x_in = mask*h ----------
__global__ __launch_bounds__(256) void mask_kernel(
    const float* __restrict__ hWm, const float* __restrict__ instr,
    const int* __restrict__ batch, const float* __restrict__ h,
    float* __restrict__ mask, float* __restrict__ x_in,
    int N, int C, float inv_sqrt_c)
{
    int wv = threadIdx.x >> 5, lane = threadIdx.x & 31;
    int n = blockIdx.x * 8 + wv;
    if (n >= N) return;
    int g = batch[n];
    const float* hw = hWm + (size_t)n * C;
    const float* iv = instr + (size_t)g * C;
    float p = 0.f;
    for (int c = lane; c < C; c += 32) p += hw[c] * iv[c];
    p = wave_sum32(p);
    float m = 1.f / (1.f + __expf(-p * inv_sqrt_c));
    if (lane == 0) mask[n] = m;
    const float* hr = h + (size_t)n * C;
    float* xo = x_in + (size_t)n * C;
    for (int c = lane; c < C; c += 32) xo[c] = m * hr[c];
}

// ---------- fused edge scores:  e[E,H] = <leaky(xl[src]+xr[dst]+edge_attr@We), att> ----------
// 64 edges per block; edge_attr tile staged via async global->LDS (f32); We fragments from
// global (L2-resident); WMMA tiles consumed in registers (ea [E,HC] never materialized).
#define LDAE_F32 132
__global__ __launch_bounds__(256) void edge_score_kernel(
    const float* __restrict__ edge_attr, const float* __restrict__ We,
    const float* __restrict__ xl, const float* __restrict__ xr,
    const float* __restrict__ att, const int* __restrict__ ei,
    float* __restrict__ escore, int E, int C, int HC, int H)
{
    __shared__ float sAf[64 * LDAE_F32];
    __shared__ float sScore[64 * 8];   // padded head stride 8
    const int tid = threadIdx.x;
    const int lane = tid & 31, wv = tid >> 5;
    const int wm = wv & 3, wn = wv >> 2;
    const int eb = blockIdx.x * 64;
    const int lane15 = lane & 15, hi = lane >> 4, kb = hi * 8;

    { // async-stage edge_attr tile [64 x C] (f32, raw bytes)
        int r = tid >> 2, part = tid & 3;
        int ge = eb + r;
        int cbase = part * (C >> 2);
        if (ge < E) {
            const float* ap = edge_attr + (size_t)ge * C + cbase;
            float* d = &sAf[r * LDAE_F32 + cbase];
            for (int j = 0; j < (C >> 4); j++) async_b128(d + j * 4, ap + j * 4);
        }
    }
    for (int t = tid; t < 64 * 8; t += 256) sScore[t] = 0.f;
    wait_async0();
    __syncthreads();

    const int ntiles = HC / 16;
    for (int nt = wn; nt < ntiles; nt += 2) {
        v8f acc = {};
        const int ng = nt * 16 + lane15;
        for (int k0 = 0; k0 < C; k0 += 32) {
            v16h a, b;
            frag_from_lds_f32(a, &sAf[(wm * 16 + lane15) * LDAE_F32 + k0], kb);
#pragma unroll
            for (int j = 0; j < 8; j++) {
                b[j]     = (_Float16)We[(size_t)(k0 + kb + j) * HC + ng];
                b[j + 8] = (_Float16)We[(size_t)(k0 + 16 + kb + j) * HC + ng];
            }
            acc = __builtin_amdgcn_wmma_f32_16x16x32_f16(false, a, false, b, (short)0, acc, false, false);
        }
        // epilogue: add gathered xl/xr, leaky-relu, contract with att
        const int hh = ng / C;
        const float av = att[ng];              // att [H,C] == flat over HC
#pragma unroll
        for (int r = 0; r < 8; r++) {
            int m = hi * 8 + r;
            int e = eb + wm * 16 + m;
            float p = 0.f;
            if (e < E) {
                int s = ei[e], d = ei[E + e];
                float v = acc[r] + xl[(size_t)s * HC + ng] + xr[(size_t)d * HC + ng];
                v = (v > 0.f) ? v : NEG_SLOPE * v;
                p = v * av;
            }
            // reduce across the 16 lanes of this half (same edge row, 16 cols of one head)
            p += __shfl_xor(p, 1, 32); p += __shfl_xor(p, 2, 32);
            p += __shfl_xor(p, 4, 32); p += __shfl_xor(p, 8, 32);
            if (lane15 == 0) atomicAdd(&sScore[(wm * 16 + m) * 8 + hh], p);
        }
    }
    __syncthreads();
    for (int t = tid; t < 64 * H; t += 256) {
        int el = t / H, hh = t % H;
        int e = eb + el;
        if (e < E) escore[(size_t)e * H + hh] = sScore[el * 8 + hh];
    }
}

// ---------- segment softmax over dst ----------
__global__ void edge_max_kernel(const float* __restrict__ esc, const int* __restrict__ ei,
                                unsigned* __restrict__ mmax, int E, int H) {
    long long i = (long long)blockIdx.x * 256 + threadIdx.x;
    if (i >= (long long)E * H) return;
    int e = (int)(i / H), hh = (int)(i - (long long)e * H);
    int d = ei[E + e];
    atomicMax(&mmax[(size_t)d * H + hh], fenc(esc[i]));
}
__global__ void edge_expsum_kernel(float* __restrict__ esc, const int* __restrict__ ei,
                                   const unsigned* __restrict__ mmax, float* __restrict__ msum,
                                   int E, int H) {
    long long i = (long long)blockIdx.x * 256 + threadIdx.x;
    if (i >= (long long)E * H) return;
    int e = (int)(i / H), hh = (int)(i - (long long)e * H);
    int d = ei[E + e];
    float ex = __expf(esc[i] - fdec(mmax[(size_t)d * H + hh]));
    esc[i] = ex;
    atomicAdd(&msum[(size_t)d * H + hh], ex);
}
__global__ void edge_agg_kernel(const float* __restrict__ esc, const float* __restrict__ msum,
                                const int* __restrict__ ei, const float* __restrict__ xl,
                                float* __restrict__ agg, int E, int HC, int C, int H) {
    long long i = (long long)blockIdx.x * 256 + threadIdx.x;
    int cpg = HC >> 2;
    if (i >= (long long)E * cpg) return;
    int e = (int)(i / cpg);
    int c = (int)(i - (long long)e * cpg) * 4;
    int hh = c / C;
    int s = ei[e], d = ei[E + e];
    float alpha = esc[(size_t)e * H + hh] / (msum[(size_t)d * H + hh] + 1e-16f);
    float4 v = *(const float4*)(xl + (size_t)s * HC + c);
    float* out = agg + (size_t)d * HC + c;
    atomicAdd(out + 0, alpha * v.x); atomicAdd(out + 1, alpha * v.y);
    atomicAdd(out + 2, alpha * v.z); atomicAdd(out + 3, alpha * v.w);
}

// ---------- per-graph attention gate + GraphNorm ----------
__global__ __launch_bounds__(256) void node_score_kernel(
    const float* __restrict__ cr, const float* __restrict__ instr,
    const int* __restrict__ batch, float* __restrict__ sbuf,
    unsigned* __restrict__ gmax, int N, int C, float inv_sqrt_c)
{
    int wv = threadIdx.x >> 5, lane = threadIdx.x & 31;
    int n = blockIdx.x * 8 + wv;
    if (n >= N) return;
    int g = batch[n];
    const float* cv = cr + (size_t)n * C;
    const float* iv = instr + (size_t)g * C;
    float p = 0.f;
    for (int c = lane; c < C; c += 32) p += cv[c] * iv[c];
    p = wave_sum32(p) * inv_sqrt_c;
    if (lane == 0) { sbuf[n] = p; atomicMax(&gmax[g], fenc(p)); }
}
__global__ void node_exp_kernel(const float* __restrict__ sbuf, const int* __restrict__ batch,
                                const unsigned* __restrict__ gmax, float* __restrict__ sexp,
                                float* __restrict__ gsum, float* __restrict__ gcnt, int N) {
    int n = blockIdx.x * 256 + threadIdx.x;
    if (n >= N) return;
    int g = batch[n];
    float ex = __expf(sbuf[n] - fdec(gmax[g]));
    sexp[n] = ex;
    atomicAdd(&gsum[g], ex);
    atomicAdd(&gcnt[g], 1.f);
}
__global__ void node_apply_kernel(float* __restrict__ cr, const float* __restrict__ sexp,
                                  const float* __restrict__ gsum, const int* __restrict__ batch,
                                  float* __restrict__ gmean, int N, int C) {
    long long i = (long long)blockIdx.x * 256 + threadIdx.x;
    int cpg = C >> 2;
    if (i >= (long long)N * cpg) return;
    int n = (int)(i / cpg);
    int c = (int)(i - (long long)n * cpg) * 4;
    int g = batch[n];
    float a = sexp[n] / (gsum[g] + 1e-16f);
    float4 v = *(float4*)(cr + (size_t)n * C + c);
    v.x *= a; v.y *= a; v.z *= a; v.w *= a;
    *(float4*)(cr + (size_t)n * C + c) = v;
    float* gm = gmean + (size_t)g * C + c;
    atomicAdd(gm + 0, v.x); atomicAdd(gm + 1, v.y);
    atomicAdd(gm + 2, v.z); atomicAdd(gm + 3, v.w);
}
__global__ void node_subvar_kernel(const float* __restrict__ cr, const float* __restrict__ gmean,
                                   const float* __restrict__ gcnt, const float* __restrict__ gna,
                                   const int* __restrict__ batch, float* __restrict__ sub,
                                   float* __restrict__ gvar, int N, int C) {
    long long i = (long long)blockIdx.x * 256 + threadIdx.x;
    int cpg = C >> 2;
    if (i >= (long long)N * cpg) return;
    int n = (int)(i / cpg);
    int c = (int)(i - (long long)n * cpg) * 4;
    int g = batch[n];
    float cnt = fmaxf(gcnt[g], 1.f);
    float4 v = *(const float4*)(cr + (size_t)n * C + c);
    const float* gm = gmean + (size_t)g * C + c;
    float s0 = v.x - gna[c + 0] * (gm[0] / cnt);
    float s1 = v.y - gna[c + 1] * (gm[1] / cnt);
    float s2 = v.z - gna[c + 2] * (gm[2] / cnt);
    float s3 = v.w - gna[c + 3] * (gm[3] / cnt);
    *(float4*)(sub + (size_t)n * C + c) = make_float4(s0, s1, s2, s3);
    float* gv = gvar + (size_t)g * C + c;
    atomicAdd(gv + 0, s0 * s0); atomicAdd(gv + 1, s1 * s1);
    atomicAdd(gv + 2, s2 * s2); atomicAdd(gv + 3, s3 * s3);
}
__global__ void node_final_kernel(const float* __restrict__ sub, const float* __restrict__ gvar,
                                  const float* __restrict__ gcnt, const float* __restrict__ gng,
                                  const float* __restrict__ gnb, const int* __restrict__ batch,
                                  const float* __restrict__ mask, float* __restrict__ h,
                                  int N, int C) {
    long long i = (long long)blockIdx.x * 256 + threadIdx.x;
    int cpg = C >> 2;
    if (i >= (long long)N * cpg) return;
    int n = (int)(i / cpg);
    int c = (int)(i - (long long)n * cpg) * 4;
    int g = batch[n];
    float cnt = fmaxf(gcnt[g], 1.f);
    float m = mask[n];
    const float* gv = gvar + (size_t)g * C + c;
    const float4 sb = *(const float4*)(sub + (size_t)n * C + c);
    float4 hv = *(float4*)(h + (size_t)n * C + c);
    float cr0 = gng[c + 0] * sb.x * rsqrtf(gv[0] / cnt + EPS_GN) + gnb[c + 0];
    float cr1 = gng[c + 1] * sb.y * rsqrtf(gv[1] / cnt + EPS_GN) + gnb[c + 1];
    float cr2 = gng[c + 2] * sb.z * rsqrtf(gv[2] / cnt + EPS_GN) + gnb[c + 2];
    float cr3 = gng[c + 3] * sb.w * rsqrtf(gv[3] / cnt + EPS_GN) + gnb[c + 3];
    hv.x = m * (cr0 + hv.x); hv.y = m * (cr1 + hv.y);
    hv.z = m * (cr2 + hv.z); hv.w = m * (cr3 + hv.w);
    *(float4*)(h + (size_t)n * C + c) = hv;
}

// ---------------- host ----------------
static inline int gdiv(long long a, long long b) { return (int)((a + b - 1) / b); }

extern "C" void kernel_launch(void* const* d_in, const int* in_sizes, int n_in,
                              void* d_out, int out_size, void* d_ws, size_t ws_size,
                              hipStream_t stream) {
    const float* x         = (const float*)d_in[0];
    const int*   ei        = (const int*)d_in[1];
    const float* instr     = (const float*)d_in[2];
    const float* edge_attr = (const float*)d_in[4];
    const int*   batch     = (const int*)d_in[5];
    const float* Wl  = (const float*)d_in[6];
    const float* Wr  = (const float*)d_in[7];
    const float* We  = (const float*)d_in[8];
    const float* att = (const float*)d_in[9];
    const float* Wm  = (const float*)d_in[10];
    const float* p1w = (const float*)d_in[11];
    const float* p1b = (const float*)d_in[12];
    const float* p2w = (const float*)d_in[13];
    const float* p2b = (const float*)d_in[14];
    const float* gng = (const float*)d_in[15];
    const float* gnb = (const float*)d_in[16];
    const float* gna = (const float*)d_in[17];

    const int E   = in_sizes[1] / 2;
    const int N   = in_sizes[5];
    const int C   = in_sizes[0] / N;
    const int L   = in_sizes[14] / C;                       // p2_b [L,C]
    const int HC  = (int)((long long)in_sizes[6] / ((long long)L * C));
    const int H   = HC / C;
    const int MID = in_sizes[12] / L;                       // p1_b [L,MID]
    const float inv_sqrt_c = 1.0f / sqrtf((float)C);

    // workspace layout
    char* ws = (char*)d_ws;
    size_t off = 0;
    auto alloc = [&](size_t bytes) -> void* {
        void* p = ws + off;
        off += (bytes + 255) & ~(size_t)255;
        return p;
    };
    float*    h_buf = (float*)alloc((size_t)N * C * 4);
    float*    maskb = (float*)alloc((size_t)N * 4);
    float*    x_in  = (float*)alloc((size_t)N * C * 4);     // reused as `sub` late in layer
    float*    xl    = (float*)alloc((size_t)N * HC * 4);
    float*    xr    = (float*)alloc((size_t)N * HC * 4);
    float*    esc   = (float*)alloc((size_t)E * H * 4);
    unsigned* mmax  = (unsigned*)alloc((size_t)N * H * 4);
    float*    msum  = (float*)alloc((size_t)N * H * 4);
    float*    agg   = (float*)alloc((size_t)N * HC * 4);
    float*    cr1   = (float*)alloc((size_t)N * MID * 4);   // also hWm scratch (MID >= C)
    float*    cr    = (float*)alloc((size_t)N * C * 4);
    float*    sbuf  = (float*)alloc((size_t)N * 4);
    float*    sexp  = (float*)alloc((size_t)N * 4);
    const int G     = in_sizes[3] / C;
    unsigned* gmax  = (unsigned*)alloc((size_t)G * 4);
    float*    gsum  = (float*)alloc((size_t)G * 4);
    float*    gcnt  = (float*)alloc((size_t)G * 4);
    float*    gmean = (float*)alloc((size_t)G * C * 4);
    float*    gvar  = (float*)alloc((size_t)G * C * 4);
    (void)ws_size; (void)n_in; (void)out_size;

    hipMemcpyAsync(h_buf, x, (size_t)N * C * 4, hipMemcpyDeviceToDevice, stream);

    const dim3 blk(256);
    for (int i = 0; i < L; i++) {
        const float* Wm_i  = Wm  + (size_t)i * C * C;
        const float* Wl_i  = Wl  + (size_t)i * C * HC;
        const float* Wr_i  = Wr  + (size_t)i * C * HC;
        const float* We_i  = We  + (size_t)i * C * HC;
        const float* att_i = att + (size_t)i * H * C;
        const float* in_i  = instr + (size_t)i * G * C;
        const float* p1w_i = p1w + (size_t)i * HC * MID;
        const float* p1b_i = p1b + (size_t)i * MID;
        const float* p2w_i = p2w + (size_t)i * MID * C;
        const float* p2b_i = p2b + (size_t)i * C;
        const float* gng_i = gng + (size_t)i * C;
        const float* gnb_i = gnb + (size_t)i * C;
        const float* gna_i = gna + (size_t)i * C;

        // 1) hWm = h @ Wm  (into cr1 scratch)
        gemm_wmma<0><<<dim3(gdiv(N, 64), C / 64), blk, 0, stream>>>(h_buf, Wm_i, nullptr, cr1, N, C, C);
        // 2) mask + x_in
        mask_kernel<<<gdiv(N, 8), blk, 0, stream>>>(cr1, in_i, batch, h_buf, maskb, x_in, N, C, inv_sqrt_c);
        // 3) xl / xr
        gemm_wmma<0><<<dim3(gdiv(N, 64), HC / 64), blk, 0, stream>>>(x_in, Wl_i, nullptr, xl, N, C, HC);
        gemm_wmma<0><<<dim3(gdiv(N, 64), HC / 64), blk, 0, stream>>>(x_in, Wr_i, nullptr, xr, N, C, HC);
        // 4) zero accumulators
        zero_kernel<<<gdiv((long long)N * H, 256), blk, 0, stream>>>(mmax, (long long)N * H);
        zero_kernel<<<gdiv((long long)N * H, 256), blk, 0, stream>>>((unsigned*)msum, (long long)N * H);
        zero_kernel<<<gdiv((long long)N * HC, 256), blk, 0, stream>>>((unsigned*)agg, (long long)N * HC);
        zero_kernel<<<1, blk, 0, stream>>>(gmax, G);
        zero_kernel<<<1, blk, 0, stream>>>((unsigned*)gsum, G);
        zero_kernel<<<1, blk, 0, stream>>>((unsigned*)gcnt, G);
        zero_kernel<<<gdiv((long long)G * C, 256), blk, 0, stream>>>((unsigned*)gmean, (long long)G * C);
        zero_kernel<<<gdiv((long long)G * C, 256), blk, 0, stream>>>((unsigned*)gvar, (long long)G * C);
        // 5) fused edge scores (ea GEMM + gather + leaky + att-contract)
        edge_score_kernel<<<gdiv(E, 64), blk, 0, stream>>>(edge_attr, We_i, xl, xr, att_i, ei, esc, E, C, HC, H);
        // 6) segment softmax over dst + weighted scatter
        edge_max_kernel<<<gdiv((long long)E * H, 256), blk, 0, stream>>>(esc, ei, mmax, E, H);
        edge_expsum_kernel<<<gdiv((long long)E * H, 256), blk, 0, stream>>>(esc, ei, mmax, msum, E, H);
        edge_agg_kernel<<<gdiv((long long)E * (HC / 4), 256), blk, 0, stream>>>(esc, msum, ei, xl, agg, E, HC, C, H);
        // 7) MLP (gelu epilogues)
        gemm_wmma<1><<<dim3(gdiv(N, 64), MID / 64), blk, 0, stream>>>(agg, p1w_i, p1b_i, cr1, N, HC, MID);
        gemm_wmma<1><<<dim3(gdiv(N, 64), C / 64), blk, 0, stream>>>(cr1, p2w_i, p2b_i, cr, N, MID, C);
        // 8) per-graph attention gate + GraphNorm + residual*mask
        node_score_kernel<<<gdiv(N, 8), blk, 0, stream>>>(cr, in_i, batch, sbuf, gmax, N, C, inv_sqrt_c);
        node_exp_kernel<<<gdiv(N, 256), blk, 0, stream>>>(sbuf, batch, gmax, sexp, gsum, gcnt, N);
        node_apply_kernel<<<gdiv((long long)N * (C / 4), 256), blk, 0, stream>>>(cr, sexp, gsum, batch, gmean, N, C);
        node_subvar_kernel<<<gdiv((long long)N * (C / 4), 256), blk, 0, stream>>>(cr, gmean, gcnt, gna_i, batch, x_in, gvar, N, C);
        node_final_kernel<<<gdiv((long long)N * (C / 4), 256), blk, 0, stream>>>(x_in, gvar, gcnt, gng_i, gnb_i, batch, maskb, h_buf, N, C);
    }
    hipMemcpyAsync(d_out, h_buf, (size_t)N * C * 4, hipMemcpyDeviceToDevice, stream);
}
